// GRUBase_61108794688034
// MI455X (gfx1250) — matrix operand order repaired and checked by reading, not compiled, not run on hardware
//
#include <hip/hip_runtime.h>
#include <stdint.h>

// GRU: B=4096, T=1024, IN=5, H=16, L=4, OUT=150
#define Bsz  4096
#define Tln  1024
#define INd  5
#define OUTd 150

typedef __attribute__((ext_vector_type(16))) _Float16 v16h;
typedef __attribute__((ext_vector_type(8)))  float    v8f;

union Frag { v16h v; uint32_t u[8]; };
union V8   { v8f  v; float    f[8]; };

struct P {
    const float* x;
    const float* Wih[4]; const float* Whh[4];
    const float* bih[4]; const float* bhh[4];
    const float* fcw; const float* fcb;
    float* out;
};

static __device__ __forceinline__ uint32_t pack2h(float a, float b) {
    union { _Float16 h[2]; uint32_t u; } r;
    r.h[0] = (_Float16)a; r.h[1] = (_Float16)b;
    return r.u;
}

// Gate weights/biases are pre-scaled by -log2(e) (r,z) and -2*log2(e) (n),
// so the nonlinearities need only exp2 + add + rcp (all single VALU/trans ops):
//   sigmoid(x)  = rcp(1 + 2^(-log2e * x))
//   tanh(x)     = 2*rcp(1 + 2^(-2*log2e * x)) - 1
static __device__ __forceinline__ float sig_pre(float a) {   // a = -log2e * preact
    return __builtin_amdgcn_rcpf(1.0f + __builtin_amdgcn_exp2f(a));
}
static __device__ __forceinline__ float tanh_pre(float a) {  // a = -2*log2e * preact
    return 2.0f * __builtin_amdgcn_rcpf(1.0f + __builtin_amdgcn_exp2f(a)) - 1.0f;
}

#define NLOG2E  (-1.4426950408889634f)      // -log2(e)
#define N2LOG2E (-2.8853900817779268f)      // -2*log2(e)

// LDS map (bytes). Row stride 48 (32B data + 16B pad) for bank spread.
//  XBUF(l,slot) : (l*2+slot)*768, l in 0..2   -> [0, 4608)   layer l -> l+1 handoff (f16, double buffered)
//  HAREA(l)     : 4608 + l*768                -> [4608, 7680) per-layer h feedback (f16)
//  HIDDEN       : 7680, float[16][64]         -> [7680, 11776) final states for FC
#define LDS_BYTES 11776

__launch_bounds__(128)
__global__ void gru_pipeline(P p) {
    __shared__ uint8_t smem[LDS_BYTES];
    const int tid   = threadIdx.x;
    const int lane  = tid & 31;
    const int layer = tid >> 5;        // wave index == GRU layer
    const int m     = lane & 15;       // matrix row / batch column
    const int hi    = lane >> 4;       // 0: lanes 0-15, 1: lanes 16-31
    const int bb    = blockIdx.x * 16; // batch tile base

    // ---- zero LDS (h state must start at 0) ----
    for (int i = tid * 4; i < LDS_BYTES; i += 128 * 4)
        *(uint32_t*)&smem[i] = 0u;

    // ---- build resident A fragments (stacked [W_ih | W_hh], f16, pre-scaled) ----
    // A 16x32 f16 layout: lane m holds M=m; lanes<16 carry K=0..7 & 16..23,
    // lanes>=16 carry K=8..15 & 24..31; 2 halfs packed per VGPR.
    const int d = (layer == 0) ? INd : 16;
    const float* Wih = p.Wih[layer];
    const float* Whh = p.Whh[layer];
    const int kx = hi * 8;

    Frag fR, fZ, fNX, fNH;
    for (int tile = 0; tile < 3; ++tile) {
        Frag f;
        int g = tile * 16 + m;
        float sc = (tile == 2) ? N2LOG2E : NLOG2E;   // fold exp2 conversion into weights
        #pragma unroll
        for (int i = 0; i < 4; ++i) {
            int k0 = kx + 2 * i, k1 = k0 + 1;
            f.u[i]     = pack2h(sc * ((k0 < d) ? Wih[g * d + k0] : 0.0f),
                                sc * ((k1 < d) ? Wih[g * d + k1] : 0.0f));
            f.u[4 + i] = pack2h(sc * Whh[g * 16 + k0], sc * Whh[g * 16 + k1]);
        }
        if (tile == 0) fR = f;
        else if (tile == 1) fZ = f;
        else {
            fNX = f; fNH = f;
            #pragma unroll
            for (int i = 0; i < 4; ++i) { fNX.u[4 + i] = 0u; fNH.u[i] = 0u; }
        }
    }

    // ---- bias C tiles (D-layout: lane holds N=batch, f[j] -> M = j+8*hi), pre-scaled ----
    V8 cR, cZ, cNX, cNH;
    #pragma unroll
    for (int j = 0; j < 8; ++j) {
        int gr = j + 8 * hi;
        cR.f[j]  = NLOG2E  * (p.bih[layer][gr]      + p.bhh[layer][gr]);
        cZ.f[j]  = NLOG2E  * (p.bih[layer][16 + gr] + p.bhh[layer][16 + gr]);
        cNX.f[j] = N2LOG2E * p.bih[layer][32 + gr];   // b_in
        cNH.f[j] = N2LOG2E * p.bhh[layer][32 + gr];   // b_hn
    }

    float ho[8];
    #pragma unroll
    for (int j = 0; j < 8; ++j) ho[j] = 0.0f;

    __syncthreads();

    // ---- software-pipelined scan: wave `layer` runs t = ph - layer ----
    for (int ph = 0; ph < Tln + 3; ++ph) {
        int t = ph - layer;                 // wave-uniform -> EXEC all-1s at runtime
        if (t >= 0 && t < Tln) {
            // Build B fragment [x_t ; h_{t-1}] (K=32 x N=16, f16):
            // lanes<16 carry K=0..7 (x) & 16..23 (h), lanes>=16 carry K=8..15 (x) & 24..31 (h).
            Frag bf;
            int haddr = 4608 + layer * 768 + m * 48;
            if (layer == 0) {
                const float* xp = p.x + ((size_t)(bb + m) * Tln + (size_t)t) * INd;
                uint32_t xa0 = pack2h(xp[0], xp[1]);
                uint32_t xa1 = pack2h(xp[2], xp[3]);
                uint32_t xa2 = pack2h(xp[4], 0.0f);
                int4 l4 = *(const int4*)&smem[haddr];
                int4 h4 = *(const int4*)&smem[haddr + 16];
                uint32_t hb[8] = { 0u, 0u, 0u, 0u,
                                   (uint32_t)h4.x, (uint32_t)h4.y, (uint32_t)h4.z, (uint32_t)h4.w };
                uint32_t xa[8] = { xa0, xa1, xa2, 0u, (uint32_t)l4.x, (uint32_t)l4.y, (uint32_t)l4.z, (uint32_t)l4.w };
                #pragma unroll
                for (int j = 0; j < 8; ++j) bf.u[j] = hi ? hb[j] : xa[j];
            } else {
                int rslot = (ph + 1) & 1;   // slot written by producer at ph-1
                int xaddr = ((layer - 1) * 2 + rslot) * 768 + m * 48;
                int4 l4 = *(const int4*)&smem[xaddr + hi * 16];
                int4 h4 = *(const int4*)&smem[haddr + hi * 16];
                bf.u[0] = l4.x; bf.u[1] = l4.y; bf.u[2] = l4.z; bf.u[3] = l4.w;
                bf.u[4] = h4.x; bf.u[5] = h4.y; bf.u[6] = h4.z; bf.u[7] = h4.w;
            }

            // scaled gates = -log2e*([Wih|Whh].[x;h] + bias)  (4x V_WMMA_F32_16X16X32_F16)
            V8 aR, aZ, aNX, aNH;
            aR.v  = __builtin_amdgcn_wmma_f32_16x16x32_f16(false, fR.v,  false, bf.v, (short)0, cR.v,  false, false);
            aZ.v  = __builtin_amdgcn_wmma_f32_16x16x32_f16(false, fZ.v,  false, bf.v, (short)0, cZ.v,  false, false);
            aNX.v = __builtin_amdgcn_wmma_f32_16x16x32_f16(false, fNX.v, false, bf.v, (short)0, cNX.v, false, false);
            aNH.v = __builtin_amdgcn_wmma_f32_16x16x32_f16(false, fNH.v, false, bf.v, (short)0, cNH.v, false, false);

            // elementwise GRU update in D-layout (lane=batch, f[j]=hidden j+8*hi)
            #pragma unroll
            for (int j = 0; j < 8; ++j) {
                float r = sig_pre(aR.f[j]);
                float z = sig_pre(aZ.f[j]);
                float n = tanh_pre(aNX.f[j] + r * aNH.f[j]);
                ho[j] = z * (ho[j] - n) + n;     // (1-z)*n + z*h
            }

            // pack h to f16, store for self-feedback and for layer+1
            int4 qv;
            qv.x = (int)pack2h(ho[0], ho[1]);
            qv.y = (int)pack2h(ho[2], ho[3]);
            qv.z = (int)pack2h(ho[4], ho[5]);
            qv.w = (int)pack2h(ho[6], ho[7]);
            int sub = hi * 16;                   // lanes<16: hidden 0..7, lanes>=16: 8..15
            *(int4*)&smem[haddr + sub] = qv;
            if (layer < 3) {
                int dstx = (layer * 2 + (ph & 1)) * 768 + m * 48 + sub;
                *(int4*)&smem[dstx] = qv;
            }
            if (t == Tln - 1) {                  // final state for the FC head
                float* hid = (float*)&smem[7680];
                #pragma unroll
                for (int j = 0; j < 8; ++j)
                    hid[m * 64 + layer * 16 + 8 * hi + j] = ho[j];
            }
        }
        __syncthreads();
    }

    // ---- FC head: out[b,o] = hidden[b,:64] . fc_w[o,:64] + fc_b[o] ----
    const float* hid = (const float*)&smem[7680];
    for (int idx = tid; idx < 16 * OUTd; idx += 128) {
        int b = idx / OUTd, o = idx - b * OUTd;
        float s = p.fcb[o];
        #pragma unroll 8
        for (int k = 0; k < 64; ++k)
            s += hid[b * 64 + k] * p.fcw[o * 64 + k];
        p.out[(size_t)(bb + b) * OUTd + o] = s;
    }
}

extern "C" void kernel_launch(void* const* d_in, const int* in_sizes, int n_in,
                              void* d_out, int out_size, void* d_ws, size_t ws_size,
                              hipStream_t stream) {
    (void)in_sizes; (void)n_in; (void)out_size; (void)d_ws; (void)ws_size;
    P p;
    p.x = (const float*)d_in[0];          // input_seq  [B,T,5]
    // d_in[1] = target_seq (unused by reference)
    for (int l = 0; l < 4; ++l) {
        p.Wih[l] = (const float*)d_in[2 + l * 4 + 0];
        p.Whh[l] = (const float*)d_in[2 + l * 4 + 1];
        p.bih[l] = (const float*)d_in[2 + l * 4 + 2];
        p.bhh[l] = (const float*)d_in[2 + l * 4 + 3];
    }
    p.fcw = (const float*)d_in[18];       // [150,64]
    p.fcb = (const float*)d_in[19];       // [150]
    p.out = (float*)d_out;

    gru_pipeline<<<dim3(Bsz / 16), dim3(128), 0, stream>>>(p);
}
